// FeatureCapNet_63677185130540
// MI455X (gfx1250) — compile-verified, matrix operand order checked by reading, not run on hardware
//
#include <hip/hip_runtime.h>
#include <hip/hip_bf16.h>
#include <cmath>

// ---------------------------------------------------------------------------
// FeatureCapNet forward on gfx1250 (CDNA5, wave32, WMMA).
// v3: same WMMA structure as v2 (channels-last f16 activations, transposed
// f16 weights, Ci-templated im2col, 16x64 tile/wave), plus:
//  - constant-offset store addressing (uhat: one base + 24-bit imm offsets)
//  - 2D grids for elementwise kernels (no 64-bit integer division)
// ---------------------------------------------------------------------------

typedef __attribute__((ext_vector_type(16))) _Float16 v16h;
typedef __attribute__((ext_vector_type(8)))  _Float16 v8h;
typedef __attribute__((ext_vector_type(8)))  float    v8f;

__device__ __forceinline__ v16h frag_cat(v8h lo, v8h hi) {
    union { v16h f; v8h r[2]; } u;
    u.r[0] = lo; u.r[1] = hi;
    return u.f;
}

// ---------------------------------------------------------------------------
// Implicit-GEMM conv via WMMA.
//  A (weights) : f16 [Mp][Kp], k' = (kh*KW+kw)*CI + ci, Kp = round32(K)
//  B (act)     : VECB ? f16 channels-last [b*Hi*Wi][CI] : f32 NCHW (c1 only)
//  C (out)     : f32 [Co][N], N = B*Ho*Wo (multiple of 64 here)
// Fragment layout (ISA): elements 0..7  <-> K = k0 + 8*laneHi + 0..7
//                        elements 8..15 <-> K = k0 + 16 + 8*laneHi + 0..7
// ---------------------------------------------------------------------------
template<int KH, int KW, int STRIDE, int PAD, int CI, bool VECB>
__global__ void conv_wmma2(const void* __restrict__ act_,
                           const _Float16* __restrict__ wt,
                           const float* __restrict__ bias,
                           float* __restrict__ out,
                           int B, int Hi, int Wi, int Co, int Ho, int Wo, int Kp)
{
    constexpr int K = CI * KH * KW;
    const int HWo = Ho * Wo;
    const int N   = B * HWo;                 // multiple of 64
    const int tn  = N >> 6;
    const int tm  = (Co + 15) >> 4;

    const int lane = threadIdx.x & 31;
    const int gw   = blockIdx.x * (blockDim.x >> 5) + (threadIdx.x >> 5);
    const int tile_n = gw % tn;
    const int tile_m = gw / tn;
    if (tile_m >= tm) return;

    const int laneLo = lane & 15;
    const int laneHi = lane >> 4;
    const int kOff   = laneHi * 8;
    const int m      = tile_m * 16 + laneLo;        // padded row: load always valid

    int bb[4], oh[4], ow[4];
#pragma unroll
    for (int s = 0; s < 4; ++s) {
        const int n = tile_n * 64 + s * 16 + laneLo;
        bb[s] = n / HWo;
        const int t = n - bb[s] * HWo;
        oh[s] = t / Wo;
        ow[s] = t - oh[s] * Wo;
    }

    v8f acc[4] = {{}, {}, {}, {}};
    const _Float16* wrow = wt + (size_t)m * Kp;

    for (int k0 = 0; k0 < K; k0 += 32) {
        const v8h ar0 = *(const v8h*)(wrow + k0 + kOff);
        const v8h ar1 = *(const v8h*)(wrow + k0 + 16 + kOff);
        const v16h af = frag_cat(ar0, ar1);
        if (k0 + 32 < K) __builtin_prefetch(wrow + k0 + 32, 0, 1);

#pragma unroll
        for (int s = 0; s < 4; ++s) {
            v16h bf;
            if constexpr (VECB) {
                const _Float16* act = (const _Float16*)act_;
                v8h br[2];
#pragma unroll
                for (int r = 0; r < 2; ++r) {
                    const int kp = k0 + r * 16 + kOff;   // run start, multiple of 8
                    const int kk = kp / CI;              // compile-time div
                    const int ci = kp - kk * CI;
                    v8h z = {};
                    if (kk < KH * KW) {
                        const int kh = kk / KW;
                        const int kw = kk - kh * KW;
                        const int ih = oh[s] * STRIDE - PAD + kh;
                        const int iw = ow[s] * STRIDE - PAD + kw;
                        if (ih >= 0 && ih < Hi && iw >= 0 && iw < Wi)
                            z = *(const v8h*)(act + ((size_t)(bb[s] * Hi + ih) * Wi + iw) * CI + ci);
                    }
                    br[r] = z;
                }
                bf = frag_cat(br[0], br[1]);
            } else {
                const float* act = (const float*)act_;
#pragma unroll
                for (int e = 0; e < 16; ++e) {
                    const int k = k0 + ((e < 8) ? 0 : 16) + kOff + (e & 7);
                    float bv = 0.f;
                    if (k < K) {
                        const int kk = k / CI;
                        const int ci = k - kk * CI;
                        const int kh = kk / KW;
                        const int kw = kk - kh * KW;
                        const int ih = oh[s] * STRIDE - PAD + kh;
                        const int iw = ow[s] * STRIDE - PAD + kw;
                        if (ih >= 0 && ih < Hi && iw >= 0 && iw < Wi)
                            bv = act[(((size_t)bb[s] * CI + ci) * Hi + ih) * Wi + iw];
                    }
                    bf[e] = (_Float16)bv;
                }
            }
            acc[s] = __builtin_amdgcn_wmma_f32_16x16x32_f16(false, af, false, bf,
                                                            (short)0, acc[s], false, false);
        }
    }

    // one base address per subtile; rows r*N apart (hoisted adds, not full chains)
    const int co0 = tile_m * 16 + laneHi * 8;
#pragma unroll
    for (int s = 0; s < 4; ++s) {
        const int n = tile_n * 64 + s * 16 + laneLo;
        float* ob = out + (size_t)co0 * N + n;
#pragma unroll
        for (int r = 0; r < 8; ++r) {
            if (co0 + r < Co) {
                float v = acc[s][r];
                if (bias) v += bias[co0 + r];
                ob[(size_t)r * N] = v;
            }
        }
    }
}

// ---------------------------------------------------------------------------
// Weight prep: OIHW f32 -> f16 [Mp][Kp] with k' = (kh*KW+kw)*Ci + ci, zero pad.
// ---------------------------------------------------------------------------
__global__ void prep_w(const float* __restrict__ w, _Float16* __restrict__ o,
                       int Co, int Ci, int KH, int KW, int Mp, int Kp)
{
    const int idx = blockIdx.x * 256 + threadIdx.x;
    if (idx >= Mp * Kp) return;
    const int m = idx / Kp;
    const int k = idx - m * Kp;
    const int K = Ci * KH * KW;
    float v = 0.f;
    if (m < Co && k < K) {
        const int kk = k / Ci;
        const int ci = k - kk * Ci;
        const int kh = kk / KW;
        const int kw = kk - kh * KW;
        v = w[(((size_t)m * Ci + ci) * KH + kh) * KW + kw];
    }
    o[idx] = (_Float16)v;
}

// route_w (14792,16,2,16) f32 -> f16 [n][jk][l]
__global__ void prep_rw(const float* __restrict__ rw, _Float16* __restrict__ o)
{
    const long i = (long)blockIdx.x * 256 + threadIdx.x;
    if (i >= (long)14792 * 512) return;
    const long n  = i >> 9;
    const int rem = (int)(i & 511);
    const int jk  = rem >> 4;
    const int l   = rem & 15;
    o[i] = (_Float16)rw[(n * 16 + l) * 32 + jk];
}

// ---------------------------------------------------------------------------
// BatchNorm stats over per-channel-contiguous f32 [C][N].
// ---------------------------------------------------------------------------
__global__ void bn_stats2(const float* __restrict__ x, int N,
                          float* __restrict__ mean, float* __restrict__ rstd, float eps)
{
    const int c = blockIdx.x;
    const int tid = threadIdx.x;
    const float* xc = x + (size_t)c * N;
    float s = 0.f, s2 = 0.f;
    for (int i = tid; i < N; i += 256) {
        const float v = xc[i];
        s += v; s2 += v * v;
    }
    __shared__ float sh[256], sh2[256];
    sh[tid] = s; sh2[tid] = s2;
    __syncthreads();
    for (int st = 128; st > 0; st >>= 1) {
        if (tid < st) { sh[tid] += sh[tid + st]; sh2[tid] += sh2[tid + st]; }
        __syncthreads();
    }
    if (tid == 0) {
        const float m   = sh[0] / (float)N;
        const float var = sh2[0] / (float)N - m * m;
        mean[c] = m;
        rstd[c] = rsqrtf(var + eps);
    }
}

// BN apply + optional relu: f32 [C][N] -> f16 channels-last [N][C]
// 2D grid: blockIdx.y = channel (no integer division).
__global__ void bn_apply_h(const float* __restrict__ x, _Float16* __restrict__ y,
                           const float* mean, const float* rstd,
                           const float* g, const float* bta,
                           int C, int N, int relu)
{
    const int n = blockIdx.x * 256 + threadIdx.x;
    const int c = blockIdx.y;
    if (n >= N) return;
    float v = (x[(size_t)c * N + n] - mean[c]) * rstd[c] * g[c] + bta[c];
    if (relu) v = fmaxf(v, 0.f);
    y[(size_t)n * C + c] = (_Float16)v;
}

// residual merge: bn(xa)+bn(xb) -> f16 channels-last
__global__ void bn_add_h(const float* __restrict__ xa, const float* __restrict__ xb,
                         _Float16* __restrict__ y,
                         const float* ma, const float* ra, const float* ga, const float* ba,
                         const float* mb, const float* rb, const float* gb, const float* bb,
                         int C, int N)
{
    const int n = blockIdx.x * 256 + threadIdx.x;
    const int c = blockIdx.y;
    if (n >= N) return;
    const float va = (xa[(size_t)c * N + n] - ma[c]) * ra[c] * ga[c] + ba[c];
    const float vb = (xb[(size_t)c * N + n] - mb[c]) * rb[c] * gb[c] + bb[c];
    y[(size_t)n * C + c] = (_Float16)(va + vb);
}

// f32 [C][N] -> f16 channels-last [N][C] (pc output relayout; bias already added)
__global__ void chw_to_nhwc_h(const float* __restrict__ x, _Float16* __restrict__ y,
                              int C, int N)
{
    const int n = blockIdx.x * 256 + threadIdx.x;
    const int c = blockIdx.y;
    if (n >= N) return;
    y[(size_t)n * C + c] = (_Float16)x[(size_t)c * N + n];
}

// ---------------------------------------------------------------------------
// u_hat[b,n,jk] = sum_l u[b,n,l] * route_w[n,l,jk]; K=16 padded to 32.
// One wave per (n, batch-tile, jk-tile): 2 x b128 loads + 1 WMMA.
// Output rows are NCAP*32 elements apart -> constant 24-bit store offsets.
// ---------------------------------------------------------------------------
__global__ void uhat_wmma2(const _Float16* __restrict__ pch,
                           const _Float16* __restrict__ wr,
                           _Float16* __restrict__ uhat)
{
    constexpr int NCAP = 14792;
    constexpr int HW = 1849;
    const int lane = threadIdx.x & 31;
    const long gw = (long)blockIdx.x * (blockDim.x >> 5) + (threadIdx.x >> 5);
    if (gw >= (long)NCAP * 8 * 2) return;

    const int  jkt = (int)(gw & 1);
    const long t   = gw >> 1;
    const int  bt  = (int)(t & 7);
    const int  n   = (int)(t >> 3);

    const int laneLo = lane & 15;
    const int laneHi = lane >> 4;
    const int cap = n & 7;
    const int hw  = n >> 3;
    const int brow = bt * 16 + laneLo;
    const int jk   = jkt * 16 + laneLo;

    const v8h z  = {};
    const v8h a0 = *(const v8h*)(pch + ((size_t)brow * HW + hw) * 128 + cap * 16 + laneHi * 8);
    const v8h b0 = *(const v8h*)(wr + ((size_t)n * 32 + jk) * 16 + laneHi * 8);
    const v16h af = frag_cat(a0, z);
    const v16h bf = frag_cat(b0, z);

    v8f acc = {};
    acc = __builtin_amdgcn_wmma_f32_16x16x32_f16(false, af, false, bf,
                                                 (short)0, acc, false, false);

    // base row = bt*16 + laneHi*8; rows r are a constant NCAP*32 elements apart
    _Float16* ub = uhat + ((size_t)(bt * 16 + laneHi * 8) * NCAP + n) * 32 + jk;
#pragma unroll
    for (int r = 0; r < 8; ++r)
        ub[(size_t)r * (NCAP * 32)] = (_Float16)acc[r];
}

// ---------------------------------------------------------------------------
// Routing: s[b,j,k] = sum_n softmax_j(blog) * u_hat ; v = squash(s).
// ---------------------------------------------------------------------------
__global__ void route_reduce(const _Float16* __restrict__ uhat,
                             const float* __restrict__ blog,
                             float* __restrict__ vout, int first)
{
    const int NCAP = 14792;
    const int b   = blockIdx.x;
    const int tid = threadIdx.x;
    float s[32];
#pragma unroll
    for (int i = 0; i < 32; ++i) s[i] = 0.f;

    for (int n = tid; n < NCAP; n += 256) {
        float c0 = 0.5f, c1 = 0.5f;
        if (!first) {
            const float b0 = blog[((size_t)b * NCAP + n) * 2 + 0];
            const float b1 = blog[((size_t)b * NCAP + n) * 2 + 1];
            const float mx = fmaxf(b0, b1);
            const float e0 = __expf(b0 - mx), e1 = __expf(b1 - mx);
            const float inv = 1.f / (e0 + e1);
            c0 = e0 * inv; c1 = e1 * inv;
        }
        const v8h* u = (const v8h*)(uhat + ((size_t)b * NCAP + n) * 32);
        const v8h u0 = u[0], u1 = u[1], u2 = u[2], u3 = u[3];
#pragma unroll
        for (int k = 0; k < 8; ++k) {
            s[k]      += c0 * (float)u0[k];
            s[8 + k]  += c0 * (float)u1[k];
            s[16 + k] += c1 * (float)u2[k];
            s[24 + k] += c1 * (float)u3[k];
        }
    }

    __shared__ float sh[256];
    __shared__ float stot[32];
    for (int comp = 0; comp < 32; ++comp) {
        sh[tid] = s[comp];
        __syncthreads();
        for (int st = 128; st > 0; st >>= 1) {
            if (tid < st) sh[tid] += sh[tid + st];
            __syncthreads();
        }
        if (tid == 0) stot[comp] = sh[0];
        __syncthreads();
    }

    if (tid < 2) {
        const int j = tid;
        float n2 = 0.f;
        for (int k = 0; k < 16; ++k) { const float x = stot[j * 16 + k]; n2 += x * x; }
        const float scale = n2 / (1.f + n2) * rsqrtf(n2 + 1e-6f);
        for (int k = 0; k < 16; ++k)
            vout[((size_t)b * 2 + j) * 16 + k] = stot[j * 16 + k] * scale;
    }
}

// blog[b,n,j] (+)= sum_k u_hat[b,n,j,k] * v[b,j,k]; 2D grid (y = batch).
__global__ void route_update(const _Float16* __restrict__ uhat,
                             const float* __restrict__ v,
                             float* __restrict__ blog, int first)
{
    const int NCAP = 14792;
    const int n = blockIdx.x * 256 + threadIdx.x;
    const int b = blockIdx.y;
    if (n >= NCAP) return;
    const size_t i = (size_t)b * NCAP + n;
    const v8h* u = (const v8h*)(uhat + i * 32);
    const v8h u0 = u[0], u1 = u[1], u2 = u[2], u3 = u[3];
    const float* vb = v + (size_t)b * 32;
    float d0 = 0.f, d1 = 0.f;
#pragma unroll
    for (int k = 0; k < 8; ++k) {
        d0 += (float)u0[k] * vb[k];
        d0 += (float)u1[k] * vb[8 + k];
        d1 += (float)u2[k] * vb[16 + k];
        d1 += (float)u3[k] * vb[24 + k];
    }
    blog[i * 2 + 0] = (first ? 0.f : blog[i * 2 + 0]) + d0;
    blog[i * 2 + 1] = (first ? 0.f : blog[i * 2 + 1]) + d1;
}

// ---------------------------------------------------------------------------
// Host orchestration
// ---------------------------------------------------------------------------
extern "C" void kernel_launch(void* const* d_in, const int* in_sizes, int n_in,
                              void* d_out, int out_size, void* d_ws, size_t ws_size,
                              hipStream_t stream) {
    (void)in_sizes; (void)n_in; (void)out_size; (void)ws_size;

    const float* x    = (const float*)d_in[0];
    const float* c1_w = (const float*)d_in[1];
    const float* c1_g = (const float*)d_in[2];
    const float* c1_b = (const float*)d_in[3];
    const float* pc_w = (const float*)d_in[28];
    const float* pc_b = (const float*)d_in[29];
    const float* rwp  = (const float*)d_in[30];

    const int B  = 128, H1 = 47, W1 = 47;
    const int S1 = B * H1 * W1;   // 282752 (mult of 64)
    const int S2 = B * 43 * 43;   // 236672 (mult of 64)

    // ---- workspace arena (~445 MB) ----
    char* base = (char*)d_ws;
    size_t off = 0;
    auto take = [&](size_t bytes) -> char* {
        char* p = base + off;
        off += (bytes + 255) & ~(size_t)255;
        return p;
    };
    _Float16* X1h  = (_Float16*)take((size_t)S1 * 32 * 2);
    _Float16* X2h  = (_Float16*)take((size_t)S1 * 64 * 2);
    _Float16* X3h  = (_Float16*)take((size_t)S1 * 96 * 2);
    float*    BLOG = (float*)take((size_t)B * 14792 * 2 * 4);
    _Float16* WR   = (_Float16*)take((size_t)14792 * 512 * 2);
    float*    V    = (float*)take((size_t)B * 32 * 4);
    float*    ST   = (float*)take(4096);
    char*     WTS  = take(1048576);
    char*     TMP  = take((size_t)304000000);

    float* MA = ST;        float* RA = ST + 128;
    float* MB = ST + 256;  float* RB = ST + 384;

    float*    P1  = (float*)(TMP);                 // mid pre-BN a      (<=27.2 MB)
    _Float16* M1h = (_Float16*)(TMP + 28000000);   // mid act a f16     (<=13.6 MB)
    float*    P2  = (float*)(TMP + 42000000);      // mid pre-BN b      (<=27.2 MB)
    _Float16* M2h = (_Float16*)(TMP + 70000000);   // mid act b f16     (<=13.6 MB)
    float*    P3  = (float*)(TMP + 84000000);      // main pre-BN       (<=108.6 MB)
    float*    Psc = (float*)(TMP + 194000000);     // shortcut pre-BN   (<=108.6 MB)
    float*    P0  = (float*)(TMP);                 // stage-1 pre-BN    (36.2 MB)
    float*    PCf = (float*)(TMP);                 // pc conv f32       (121.2 MB)
    _Float16* PCh = (_Float16*)(TMP + 122000000);  // pc act f16        (60.6 MB)
    _Float16* UH  = (_Float16*)(TMP);              // u_hat f16         (121.2 MB, after PCf dead)

    const float eps = 1e-5f;
    auto cvblk = [](int N, int Co) {
        const long waves = (long)(N >> 6) * ((Co + 15) / 16);
        return (int)((waves + 7) / 8);
    };
    auto nblk = [](int N) { return (N + 255) / 256; };

    // ---- weight prep (f16, (kh,kw,ci)-ordered, padded) ----
    size_t woff = 0;
    auto prep = [&](const float* w, int Co, int Ci, int KH, int KW) -> _Float16* {
        const int K  = Ci * KH * KW;
        const int Kp = (K + 31) & ~31;
        const int Mp = (Co + 15) & ~15;
        _Float16* p = (_Float16*)(WTS + woff);
        woff = (woff + (size_t)Mp * Kp * 2 + 255) & ~(size_t)255;
        prep_w<<<(Mp * Kp + 255) / 256, 256, 0, stream>>>(w, p, Co, Ci, KH, KW, Mp, Kp);
        return p;
    };
    _Float16* Wc1  = prep(c1_w, 32, 3, 3, 3);
    _Float16* Wr1a = prep((const float*)d_in[4], 16, 32, 1, 1);
    _Float16* Wr1b = prep((const float*)d_in[7], 16, 16, 3, 3);
    _Float16* Wr1c = prep((const float*)d_in[10], 64, 16, 1, 1);
    _Float16* Wr1s = prep((const float*)d_in[13], 64, 32, 1, 1);
    _Float16* Wr2a = prep((const float*)d_in[16], 24, 64, 1, 1);
    _Float16* Wr2b = prep((const float*)d_in[19], 24, 24, 3, 3);
    _Float16* Wr2c = prep((const float*)d_in[22], 96, 24, 1, 1);
    _Float16* Wr2s = prep((const float*)d_in[25], 96, 64, 1, 1);
    _Float16* Wpc  = prep(pc_w, 128, 96, 5, 5);
    prep_rw<<<(int)(((long)14792 * 512 + 255) / 256), 256, 0, stream>>>(rwp, WR);

    // ---- stage 1: conv 3x3 s2 (3->32, scalar-gather B), BN+ReLU -> X1h ----
    conv_wmma2<3, 3, 2, 0, 3, false><<<cvblk(S1, 32), 256, 0, stream>>>(
        x, Wc1, nullptr, P0, B, 96, 96, 32, H1, W1, 32);
    bn_stats2<<<32, 256, 0, stream>>>(P0, S1, MA, RA, eps);
    bn_apply_h<<<dim3(nblk(S1), 32), 256, 0, stream>>>(P0, X1h, MA, RA, c1_g, c1_b, 32, S1, 1);

    // ---- res1: 32 -> (16,16) -> 64 + shortcut ----
    {
        const float *g1=(const float*)d_in[5],  *b1=(const float*)d_in[6];
        const float *g2=(const float*)d_in[8],  *b2=(const float*)d_in[9];
        const float *g3=(const float*)d_in[11], *b3=(const float*)d_in[12];
        const float *gs=(const float*)d_in[14], *bs=(const float*)d_in[15];
        conv_wmma2<1,1,1,0,32,true><<<cvblk(S1,16),256,0,stream>>>(X1h, Wr1a, nullptr, P1, B,H1,W1, 16,H1,W1, 32);
        bn_stats2<<<16,256,0,stream>>>(P1, S1, MA, RA, eps);
        bn_apply_h<<<dim3(nblk(S1),16),256,0,stream>>>(P1, M1h, MA, RA, g1, b1, 16, S1, 1);
        conv_wmma2<3,3,1,1,16,true><<<cvblk(S1,16),256,0,stream>>>(M1h, Wr1b, nullptr, P2, B,H1,W1, 16,H1,W1, 160);
        bn_stats2<<<16,256,0,stream>>>(P2, S1, MA, RA, eps);
        bn_apply_h<<<dim3(nblk(S1),16),256,0,stream>>>(P2, M2h, MA, RA, g2, b2, 16, S1, 1);
        conv_wmma2<1,1,1,0,16,true><<<cvblk(S1,64),256,0,stream>>>(M2h, Wr1c, nullptr, P3, B,H1,W1, 64,H1,W1, 32);
        conv_wmma2<1,1,1,0,32,true><<<cvblk(S1,64),256,0,stream>>>(X1h, Wr1s, nullptr, Psc, B,H1,W1, 64,H1,W1, 32);
        bn_stats2<<<64,256,0,stream>>>(P3, S1, MA, RA, eps);
        bn_stats2<<<64,256,0,stream>>>(Psc, S1, MB, RB, eps);
        bn_add_h<<<dim3(nblk(S1),64),256,0,stream>>>(P3, Psc, X2h, MA,RA,g3,b3, MB,RB,gs,bs, 64, S1);
    }

    // ---- res2: 64 -> (24,24) -> 96 + shortcut ----
    {
        const float *g1=(const float*)d_in[17], *b1=(const float*)d_in[18];
        const float *g2=(const float*)d_in[20], *b2=(const float*)d_in[21];
        const float *g3=(const float*)d_in[23], *b3=(const float*)d_in[24];
        const float *gs=(const float*)d_in[26], *bs=(const float*)d_in[27];
        conv_wmma2<1,1,1,0,64,true><<<cvblk(S1,24),256,0,stream>>>(X2h, Wr2a, nullptr, P1, B,H1,W1, 24,H1,W1, 64);
        bn_stats2<<<24,256,0,stream>>>(P1, S1, MA, RA, eps);
        bn_apply_h<<<dim3(nblk(S1),24),256,0,stream>>>(P1, M1h, MA, RA, g1, b1, 24, S1, 1);
        conv_wmma2<3,3,1,1,24,true><<<cvblk(S1,24),256,0,stream>>>(M1h, Wr2b, nullptr, P2, B,H1,W1, 24,H1,W1, 224);
        bn_stats2<<<24,256,0,stream>>>(P2, S1, MA, RA, eps);
        bn_apply_h<<<dim3(nblk(S1),24),256,0,stream>>>(P2, M2h, MA, RA, g2, b2, 24, S1, 1);
        conv_wmma2<1,1,1,0,24,true><<<cvblk(S1,96),256,0,stream>>>(M2h, Wr2c, nullptr, P3, B,H1,W1, 96,H1,W1, 32);
        conv_wmma2<1,1,1,0,64,true><<<cvblk(S1,96),256,0,stream>>>(X2h, Wr2s, nullptr, Psc, B,H1,W1, 96,H1,W1, 64);
        bn_stats2<<<96,256,0,stream>>>(P3, S1, MA, RA, eps);
        bn_stats2<<<96,256,0,stream>>>(Psc, S1, MB, RB, eps);
        bn_add_h<<<dim3(nblk(S1),96),256,0,stream>>>(P3, Psc, X3h, MA,RA,g3,b3, MB,RB,gs,bs, 96, S1);
    }

    // ---- primary capsule conv 5x5 (96->128, +bias) then relayout to f16 ----
    conv_wmma2<5,5,1,0,96,true><<<cvblk(S2,128),256,0,stream>>>(X3h, Wpc, pc_b, PCf, B,H1,W1, 128,43,43, 2400);
    chw_to_nhwc_h<<<dim3(nblk(S2),128),256,0,stream>>>(PCf, PCh, 128, S2);

    // ---- u_hat via WMMA (PCf region now reused for UH) ----
    {
        const long waves = (long)14792 * 8 * 2;
        uhat_wmma2<<<(int)((waves + 7) / 8), 256, 0, stream>>>(PCh, WR, UH);
    }

    // ---- dynamic routing (3 iterations) ----
    const dim3 ug(nblk(14792), B);
    route_reduce<<<B, 256, 0, stream>>>(UH, BLOG, V, 1);
    route_update<<<ug, 256, 0, stream>>>(UH, V, BLOG, 1);
    route_reduce<<<B, 256, 0, stream>>>(UH, BLOG, V, 0);
    route_update<<<ug, 256, 0, stream>>>(UH, V, BLOG, 0);
    route_reduce<<<B, 256, 0, stream>>>(UH, BLOG, (float*)d_out, 0);
}
